// LinearAttention_78400333021811
// MI455X (gfx1250) — matrix operand order repaired and checked by reading, not compiled, not run on hardware
//
#include <hip/hip_runtime.h>
#include <stddef.h>

#define DIM 768
#define HEADS 12
#define HEAD_DIM 64
#define SCALE_F 0.125f
#define EPS_F 1e-6f

typedef __attribute__((ext_vector_type(16))) __bf16 v16bf;
typedef __attribute__((ext_vector_type(8)))  float  v8f;

union Frag32 { uint4 q[2]; v16bf v; };
union Pack8  { uint4 q;    __bf16 e[8]; };

// ---------- fragment loaders (CDNA5 wave32 WMMA bf16 layouts) ----------
// Base pointers are precomputed per lane; k0 is a compile-time constant so
// loads fold to base + immediate offset (no inner-loop address VALU).
// A frag: lane r=lane&15 holds row m0+r; K contiguous 16 elems per half-wave.
__device__ __forceinline__ v16bf ld_a(const __bf16* __restrict__ p, int k0) {
    Frag32 f;
    f.q[0] = *reinterpret_cast<const uint4*>(p + k0);
    f.q[1] = *reinterpret_cast<const uint4*>(p + k0 + 8);
    return f.v;
}
// B frag (B[k][n] = W[n][k]): lane c holds column n0+c; K contiguous 16 elems.
__device__ __forceinline__ v16bf ld_b(const __bf16* __restrict__ p, int k0) {
    Frag32 f;
    f.q[0] = *reinterpret_cast<const uint4*>(p + k0);
    f.q[1] = *reinterpret_cast<const uint4*>(p + k0 + 8);
    return f.v;
}

// generic-index variants (used by the small kernels)
__device__ __forceinline__ v16bf load_a_frag(const __bf16* X, int ld, int m0, int k0, int lane) {
    int r    = lane & 15;
    int koff = (lane >> 4) << 4;          // 0 or 16
    return ld_a(X + (size_t)(m0 + r) * ld + koff, k0);
}
__device__ __forceinline__ v16bf load_b_wt_frag(const __bf16* W, int ld, int k0, int n0, int lane) {
    int c  = lane & 15;
    int kb = (lane >> 4) << 4;            // 0 or 16
    return ld_b(W + (size_t)(n0 + c) * ld + kb, k0);
}

__device__ __forceinline__ v8f wmma_bf16(v16bf a, v16bf b, v8f c) {
    return __builtin_amdgcn_wmma_f32_16x16x32_bf16(false, a, false, b, (short)0, c, false, false);
}

__device__ __forceinline__ float phi_elu1(float x) {
    return x > 0.f ? x + 1.f : __expf(x);
}

// ---------- shared GEMM core: 32(M) x 64(N) wave tile ----------
// Fully unrolled K with ping-pong register buffers: base pointers fixed,
// all loads use immediate offsets, loads issue one half-step ahead of use.
__device__ __forceinline__ void wmma_group(const v16bf a[2], const v16bf b[4], v8f acc[2][4]) {
#pragma unroll
    for (int t = 0; t < 4; ++t) {
        acc[0][t] = wmma_bf16(a[0], b[t], acc[0][t]);
        acc[1][t] = wmma_bf16(a[1], b[t], acc[1][t]);
    }
}

__device__ __forceinline__ void gemm_core_32x64(const __bf16* __restrict__ X,
                                                const __bf16* __restrict__ W,
                                                int m0, int n0, int lane,
                                                v8f acc[2][4]) {
    const int r    = lane & 15;
    const int koff = (lane >> 4) << 4;    // A: 0 or 16
    const int kb   = (lane >> 4) << 4;    // B: 0 or 16

    const __bf16* __restrict__ pa0 = X + (size_t)(m0 + r) * DIM + koff;
    const __bf16* __restrict__ pa1 = pa0 + (size_t)16 * DIM;
    const __bf16* __restrict__ pb0 = W + (size_t)(n0 + r) * DIM + kb;
    const __bf16* __restrict__ pb1 = pb0 + (size_t)16 * DIM;
    const __bf16* __restrict__ pb2 = pb1 + (size_t)16 * DIM;
    const __bf16* __restrict__ pb3 = pb2 + (size_t)16 * DIM;

    v16bf a0[2], b0[4], a1[2], b1[4];
    a0[0] = ld_a(pa0, 0); a0[1] = ld_a(pa1, 0);
    b0[0] = ld_b(pb0, 0); b0[1] = ld_b(pb1, 0); b0[2] = ld_b(pb2, 0); b0[3] = ld_b(pb3, 0);

#pragma unroll
    for (int k0 = 0; k0 < DIM - 64; k0 += 64) {     // 11 unrolled iterations
        a1[0] = ld_a(pa0, k0 + 32); a1[1] = ld_a(pa1, k0 + 32);
        b1[0] = ld_b(pb0, k0 + 32); b1[1] = ld_b(pb1, k0 + 32);
        b1[2] = ld_b(pb2, k0 + 32); b1[3] = ld_b(pb3, k0 + 32);
        if (k0 + 256 < DIM) {                        // global_prefetch_b8, imm offset
            __builtin_prefetch(pa0 + k0 + 256, 0, 1);
            __builtin_prefetch(pa1 + k0 + 256, 0, 1);
        }
        wmma_group(a0, b0, acc);
        a0[0] = ld_a(pa0, k0 + 64); a0[1] = ld_a(pa1, k0 + 64);
        b0[0] = ld_b(pb0, k0 + 64); b0[1] = ld_b(pb1, k0 + 64);
        b0[2] = ld_b(pb2, k0 + 64); b0[3] = ld_b(pb3, k0 + 64);
        wmma_group(a1, b1, acc);
    }
    // tail: a0/b0 hold k=DIM-64; load k=DIM-32; no further loads
    a1[0] = ld_a(pa0, DIM - 32); a1[1] = ld_a(pa1, DIM - 32);
    b1[0] = ld_b(pb0, DIM - 32); b1[1] = ld_b(pb1, DIM - 32);
    b1[2] = ld_b(pb2, DIM - 32); b1[3] = ld_b(pb3, DIM - 32);
    wmma_group(a0, b0, acc);
    wmma_group(a1, b1, acc);
}

// ---------- kernel 0: fp32 -> bf16 convert (4 elems/thread) ----------
__global__ void cvt_f32_bf16(const float* __restrict__ in, __bf16* __restrict__ out, size_t n4) {
    size_t i = (size_t)blockIdx.x * blockDim.x + threadIdx.x;
    if (i < n4) {
        float4 f = reinterpret_cast<const float4*>(in)[i];
        __bf16 b4[4] = {(__bf16)f.x, (__bf16)f.y, (__bf16)f.z, (__bf16)f.w};
        reinterpret_cast<uint2*>(out)[i] = *reinterpret_cast<uint2*>(b4);
    }
}

// ---------- kernel 1: fused Q/K/V projection GEMM + phi epilogue ----------
// grid: (M/256, DIM/64, 3)  block: 256 (8 waves). Wave computes 32x64 tile.
__global__ void proj_qkv_kernel(const __bf16* __restrict__ Xq, const __bf16* __restrict__ Xkv,
                                const __bf16* __restrict__ Wq, const __bf16* __restrict__ Wk,
                                const __bf16* __restrict__ Wv,
                                const float* __restrict__ bq, const float* __restrict__ bk,
                                const float* __restrict__ bv,
                                __bf16* __restrict__ Qo, __bf16* __restrict__ Ko,
                                __bf16* __restrict__ Vo) {
    const int mode = blockIdx.z;
    const __bf16* X = (mode == 0) ? Xq : Xkv;
    const __bf16* W = (mode == 0) ? Wq : (mode == 1 ? Wk : Wv);
    const float*  bias = (mode == 0) ? bq : (mode == 1 ? bk : bv);
    __bf16*       Out  = (mode == 0) ? Qo : (mode == 1 ? Ko : Vo);

    const int wave = threadIdx.x >> 5, lane = threadIdx.x & 31;
    const int m0 = blockIdx.x * 256 + wave * 32;
    const int n0 = blockIdx.y * 64;

    v8f acc[2][4] = {{v8f{}, v8f{}, v8f{}, v8f{}}, {v8f{}, v8f{}, v8f{}, v8f{}}};
    gemm_core_32x64(X, W, m0, n0, lane, acc);

    const int hi = (lane >> 4) << 3;
    const int c  = lane & 15;
#pragma unroll
    for (int mi = 0; mi < 2; ++mi) {
        const int rbase = m0 + mi * 16 + hi;
#pragma unroll
        for (int t = 0; t < 4; ++t) {
            const int col = n0 + t * 16 + c;
            const float bb = bias[col];
#pragma unroll
            for (int i = 0; i < 8; ++i) {
                float x = acc[mi][t][i] + bb;
                if (mode == 0)      x = phi_elu1(x * SCALE_F);
                else if (mode == 1) x = phi_elu1(x);
                Out[(size_t)(rbase + i) * DIM + col] = (__bf16)x;
            }
        }
    }
}

// ---------- kernel 2: per-head KV state + k_sum ----------
// grid: (B*HEADS)  block: 512 (16 waves). Wave w -> output tile (w/4, w%4) of 64x64.
// kv stored TRANSPOSED (kvT[e][d]) so the apply-GEMM B-frags read contiguously.
__global__ void kv_state_kernel(const __bf16* __restrict__ Kphi, const __bf16* __restrict__ Vp,
                                __bf16* __restrict__ kvT, float* __restrict__ ksum, int Nk) {
    __shared__ __attribute__((aligned(16))) __bf16 sK[64][40];  // [d][n] transposed, padded
    __shared__ __attribute__((aligned(16))) __bf16 sV[64][40];  // [e][n] transposed, padded

    const int bh = blockIdx.x;
    const int b = bh / HEADS, h = bh % HEADS;
    const __bf16* Kp = Kphi + (size_t)b * Nk * DIM + h * HEAD_DIM;
    const __bf16* Vv = Vp   + (size_t)b * Nk * DIM + h * HEAD_DIM;

    const int tid = threadIdx.x;
    const int lane = tid & 31, wave = tid >> 5;
    const int mi = (wave >> 2) * 16, ni = (wave & 3) * 16;

    v8f acc = v8f{};
    float ks = 0.f;

    for (int n0 = 0; n0 < Nk; n0 += 32) {
        __syncthreads();
        {   // stage 32x64 tiles of k_phi and v into LDS, transposed
            int t = tid;
            const __bf16* src = (t < 256) ? Kp : Vv;
            __bf16(*dst)[40] = (t < 256) ? sK : sV;
            if (t >= 256) t -= 256;
            const int row = t >> 3;          // 0..31 (sequence)
            const int cb  = (t & 7) << 3;    // 0..56 (head dim)
            Pack8 p;
            p.q = *reinterpret_cast<const uint4*>(src + (size_t)(n0 + row) * DIM + cb);
#pragma unroll
            for (int j = 0; j < 8; ++j) dst[cb + j][row] = p.e[j];
        }
        __syncthreads();

        // fused k_sum partial (threads 0..63 own one head-dim each)
        if (tid < 64) {
#pragma unroll 8
            for (int n = 0; n < 32; ++n) ks += (float)sK[tid][n];
        }

        // WMMA fragments from LDS (both contiguous per lane, 16B aligned)
        const int r = lane & 15;
        const int koff = (lane >> 4) << 4;
        const int kb   = (lane >> 4) << 4;
        Frag32 fa, fb;
        const __bf16* pa = &sK[mi + r][koff];
        fa.q[0] = *reinterpret_cast<const uint4*>(pa);
        fa.q[1] = *reinterpret_cast<const uint4*>(pa + 8);
        const __bf16* pb = &sV[ni + r][kb];
        fb.q[0] = *reinterpret_cast<const uint4*>(pb);
        fb.q[1] = *reinterpret_cast<const uint4*>(pb + 8);
        acc = wmma_bf16(fa.v, fb.v, acc);
    }

    // write kvT[e][d] (transposed) in bf16
    const int rb = mi + ((lane >> 4) << 3);   // d
    const int cc = ni + (lane & 15);          // e
    __bf16* outp = kvT + (size_t)bh * (HEAD_DIM * HEAD_DIM);
#pragma unroll
    for (int i = 0; i < 8; ++i) outp[(size_t)cc * HEAD_DIM + rb + i] = (__bf16)acc[i];
    if (tid < 64) ksum[bh * HEAD_DIM + tid] = ks;
}

// ---------- kernel 3: out = (q_phi @ kv) / (q_phi . k_sum + eps) ----------
// grid: (Nq/256, B*HEADS)  block: 256 (8 waves). Wave: 32 rows x 64 cols, K=64.
__global__ void apply_kernel(const __bf16* __restrict__ Qphi, const __bf16* __restrict__ kvT,
                             const float* __restrict__ ksum, __bf16* __restrict__ Attn, int Nq) {
    const int bh = blockIdx.y;
    const int b = bh / HEADS, h = bh % HEADS;
    const int wave = threadIdx.x >> 5, lane = threadIdx.x & 31;
    const int m0 = blockIdx.x * 256 + wave * 32;

    const __bf16* Qp = Qphi + (size_t)b * Nq * DIM + h * HEAD_DIM;
    const __bf16* KV = kvT + (size_t)bh * (HEAD_DIM * HEAD_DIM);
    const float*  Ks = ksum + bh * HEAD_DIM;

    v8f acc[2][4] = {{v8f{}, v8f{}, v8f{}, v8f{}}, {v8f{}, v8f{}, v8f{}, v8f{}}};
#pragma unroll
    for (int k0 = 0; k0 < HEAD_DIM; k0 += 32) {
        v16bf a0 = load_a_frag(Qp, DIM, m0,      k0, lane);
        v16bf a1 = load_a_frag(Qp, DIM, m0 + 16, k0, lane);
#pragma unroll
        for (int t = 0; t < 4; ++t) {
            // logical B[k][n] = kv[k][n] = kvT[n*64 + k] -> contiguous along k
            v16bf bfr = load_b_wt_frag(KV, HEAD_DIM, k0, t * 16, lane);
            acc[0][t] = wmma_bf16(a0, bfr, acc[0][t]);
            acc[1][t] = wmma_bf16(a1, bfr, acc[1][t]);
        }
    }

    // normalizer z: each of the 32 lanes owns one row m0+lane
    float z = EPS_F;
    {
        const __bf16* qrow = Qp + (size_t)(m0 + lane) * DIM;
#pragma unroll 8
        for (int k = 0; k < HEAD_DIM; ++k) z += (float)qrow[k] * Ks[k];
    }

    const int hi = (lane >> 4) << 3;
    const int c  = lane & 15;
    __bf16* Op = Attn + (size_t)b * Nq * DIM + h * HEAD_DIM;
#pragma unroll
    for (int mi = 0; mi < 2; ++mi) {
        const int rbase = m0 + mi * 16 + hi;
#pragma unroll
        for (int i = 0; i < 8; ++i) {
            const float zi = __shfl(z, mi * 16 + i + hi, 32);
            const float rz = 1.f / zi;
#pragma unroll
            for (int t = 0; t < 4; ++t)
                Op[(size_t)(rbase + i) * DIM + t * 16 + c] = (__bf16)(acc[mi][t][i] * rz);
        }
    }
}

// ---------- kernel 4: final projection, fp32 out ----------
// grid: (M/256, DIM/64)  block: 256. Wave computes 32x64 tile.
__global__ void out_proj_kernel(const __bf16* __restrict__ A, const __bf16* __restrict__ Wp,
                                const float* __restrict__ bp, float* __restrict__ Out) {
    const int wave = threadIdx.x >> 5, lane = threadIdx.x & 31;
    const int m0 = blockIdx.x * 256 + wave * 32;
    const int n0 = blockIdx.y * 64;

    v8f acc[2][4] = {{v8f{}, v8f{}, v8f{}, v8f{}}, {v8f{}, v8f{}, v8f{}, v8f{}}};
    gemm_core_32x64(A, Wp, m0, n0, lane, acc);

    const int hi = (lane >> 4) << 3;
    const int c  = lane & 15;
#pragma unroll
    for (int mi = 0; mi < 2; ++mi) {
        const int rbase = m0 + mi * 16 + hi;
#pragma unroll
        for (int t = 0; t < 4; ++t) {
            const int col = n0 + t * 16 + c;
            const float bb = bp[col];
#pragma unroll
            for (int i = 0; i < 8; ++i)
                Out[(size_t)(rbase + i) * DIM + col] = acc[mi][t][i] + bb;
        }
    }
}

// ---------- host launch ----------
extern "C" void kernel_launch(void* const* d_in, const int* in_sizes, int n_in,
                              void* d_out, int out_size, void* d_ws, size_t ws_size,
                              hipStream_t stream) {
    const int B = 4, Nq = 4096, Nk = 4096;
    const float* x_q  = (const float*)d_in[0];
    const float* x_kv = (const float*)d_in[1];
    const float* Wq = (const float*)d_in[2];
    const float* bq = (const float*)d_in[3];
    const float* Wk = (const float*)d_in[4];
    const float* bk = (const float*)d_in[5];
    const float* Wv = (const float*)d_in[6];
    const float* bv = (const float*)d_in[7];
    const float* Wp = (const float*)d_in[8];
    const float* bp = (const float*)d_in[9];
    float* out = (float*)d_out;

    const size_t nAct = (size_t)B * Nq * DIM;       // 12,582,912
    const size_t nW   = (size_t)DIM * DIM;          // 589,824
    const size_t nKV  = (size_t)B * HEADS * HEAD_DIM * HEAD_DIM;

    char* ws = (char*)d_ws;
    __bf16* xq_bf  = (__bf16*)ws;               ws += nAct * sizeof(__bf16);
    __bf16* xkv_bf = (__bf16*)ws;               ws += nAct * sizeof(__bf16);
    __bf16* qphi   = (__bf16*)ws;               ws += nAct * sizeof(__bf16);
    __bf16* kphi   = (__bf16*)ws;               ws += nAct * sizeof(__bf16);
    __bf16* vproj  = (__bf16*)ws;               ws += nAct * sizeof(__bf16);
    __bf16* wq_bf  = (__bf16*)ws;               ws += nW * sizeof(__bf16);
    __bf16* wk_bf  = (__bf16*)ws;               ws += nW * sizeof(__bf16);
    __bf16* wv_bf  = (__bf16*)ws;               ws += nW * sizeof(__bf16);
    __bf16* wp_bf  = (__bf16*)ws;               ws += nW * sizeof(__bf16);
    __bf16* kvT    = (__bf16*)ws;               ws += nKV * sizeof(__bf16);
    float*  ksum   = (float*)ws;                ws += (size_t)B * HEADS * HEAD_DIM * sizeof(float);
    __bf16* attn   = xkv_bf;  // alias: x_kv bf16 copy is dead after projections

    const int CT = 256;
    cvt_f32_bf16<<<dim3((unsigned)((nAct / 4 + CT - 1) / CT)), dim3(CT), 0, stream>>>(x_q,  xq_bf,  nAct / 4);
    cvt_f32_bf16<<<dim3((unsigned)((nAct / 4 + CT - 1) / CT)), dim3(CT), 0, stream>>>(x_kv, xkv_bf, nAct / 4);
    cvt_f32_bf16<<<dim3((unsigned)((nW / 4 + CT - 1) / CT)),   dim3(CT), 0, stream>>>(Wq, wq_bf, nW / 4);
    cvt_f32_bf16<<<dim3((unsigned)((nW / 4 + CT - 1) / CT)),   dim3(CT), 0, stream>>>(Wk, wk_bf, nW / 4);
    cvt_f32_bf16<<<dim3((unsigned)((nW / 4 + CT - 1) / CT)),   dim3(CT), 0, stream>>>(Wv, wv_bf, nW / 4);
    cvt_f32_bf16<<<dim3((unsigned)((nW / 4 + CT - 1) / CT)),   dim3(CT), 0, stream>>>(Wp, wp_bf, nW / 4);

    const int M = B * Nq;  // 16384
    proj_qkv_kernel<<<dim3(M / 256, DIM / 64, 3), dim3(256), 0, stream>>>(
        xq_bf, xkv_bf, wq_bf, wk_bf, wv_bf, bq, bk, bv, qphi, kphi, vproj);

    kv_state_kernel<<<dim3(B * HEADS), dim3(512), 0, stream>>>(kphi, vproj, kvT, ksum, Nk);

    apply_kernel<<<dim3(Nq / 256, B * HEADS), dim3(256), 0, stream>>>(qphi, kvT, ksum, attn, Nq);

    out_proj_kernel<<<dim3(M / 256, DIM / 64), dim3(256), 0, stream>>>(attn, wp_bf, bp, out);
}
